// OlmoLMHeadCE_18545668784153
// MI455X (gfx1250) — matrix operand order repaired and checked by reading, not compiled, not run on hardware
//
#include <hip/hip_runtime.h>
#include <hip/hip_bf16.h>
#include <math.h>

typedef __attribute__((ext_vector_type(16))) __bf16 v16bf;
typedef __attribute__((ext_vector_type(8)))  float  v8f;

constexpr int NROWS = 4096;
constexpr int DDIM  = 2048;
constexpr int VDIM  = 50257;
constexpr int CBW   = 128;                         // cols per stats col-block
constexpr int NUMCB = (VDIM + CBW - 1) / CBW;      // 393
constexpr int GRIDX = (NUMCB + 1) / 2;             // 197 (each WG covers 2 col-blocks = 256 cols)
constexpr int VPAD  = GRIDX * 256;                 // 50432, padded col count (zero-filled)
constexpr int MT    = 128;                         // rows per WG
constexpr int KS    = 32;                          // K per WMMA step (bf16)
constexpr int KIT   = DDIM / KS;                   // 64 k-slabs
constexpr int LPAD  = 20;                          // padded LDS row stride (dwords)
constexpr int IGNORE_INDEX = -100;

union Frag { unsigned u[8]; v16bf v; };

__device__ __forceinline__ unsigned bf16pair(float a, float b) {
    unsigned ua = __builtin_bit_cast(unsigned, a);
    unsigned ub = __builtin_bit_cast(unsigned, b);
    ua = (ua + 0x7FFFu + ((ua >> 16) & 1u)) >> 16;   // RNE fp32 -> bf16
    ub = (ub + 0x7FFFu + ((ub >> 16) & 1u)) >> 16;
    return (ua & 0xFFFFu) | (ub << 16);
}

// LDS byte offset of a generic pointer into __shared__ (LDS aperture keeps offset in addr[31:0])
__device__ __forceinline__ unsigned lds_off(const void* p) {
    return (unsigned)(size_t)p;
}
// Per-lane async global->LDS 16B copy, tracked by ASYNCcnt (GLOBAL_LOAD_ASYNC_TO_LDS_B128)
__device__ __forceinline__ void async_b128(unsigned ldst, const void* gsrc) {
    asm volatile("global_load_async_to_lds_b128 %0, %1, off" :: "v"(ldst), "v"(gsrc) : "memory");
}
__device__ __forceinline__ void wait_async0() {
    asm volatile("s_wait_asynccnt 0" ::: "memory");
}

// ---------------- phase 0: one-time repack to bf16, fragment-major ----------------
// bw[(slab*VPAD + n)*4 + q] : per (k-slab, col) 16 packed pairs = 4 uint4 (64B run)
__global__ __launch_bounds__(256)
void pack_w(const float* __restrict__ w, uint4* __restrict__ bw) {
    const int slab = blockIdx.y;
    const int n    = blockIdx.x * 256 + threadIdx.x;
    unsigned o[16];
    #pragma unroll
    for (int kk = 0; kk < 16; ++kk) {
        float w0 = 0.f, w1 = 0.f;
        if (n < VDIM) {
            size_t b = (size_t)(slab * KS + 2 * kk) * VDIM + n;
            w0 = w[b];
            w1 = w[b + VDIM];
        }
        o[kk] = bf16pair(w0, w1);
    }
    uint4* dst = bw + ((size_t)slab * VPAD + n) * 4;
    #pragma unroll
    for (int q = 0; q < 4; ++q) dst[q] = *(const uint4*)&o[q * 4];
}

// ax[(slab*NROWS + m)*4 + q]
__global__ __launch_bounds__(256)
void pack_x(const float* __restrict__ x, uint4* __restrict__ ax) {
    const int gt   = blockIdx.x * 256 + threadIdx.x;   // NROWS*KIT threads
    const int m    = gt >> 6;
    const int slab = gt & 63;
    const float* src = x + (size_t)m * DDIM + slab * KS;
    unsigned o[16];
    #pragma unroll
    for (int kk = 0; kk < 16; ++kk) o[kk] = bf16pair(src[2 * kk], src[2 * kk + 1]);
    uint4* dst = ax + ((size_t)slab * NROWS + m) * 4;
    #pragma unroll
    for (int q = 0; q < 4; ++q) dst[q] = *(const uint4*)&o[q * 4];
}

// ---------------- stagers ----------------
// Packed path: async global->LDS b128 copies (ASYNCcnt), no VGPR round-trip, no DScnt pollution.
__device__ __forceinline__ void load_packed_async(unsigned (&aQ)[MT][LPAD],
                                                  unsigned (&bQ)[256][LPAD],
                                                  const uint4* __restrict__ ax,
                                                  const uint4* __restrict__ bw,
                                                  int row0, int colB, int slab, int tid) {
    {
        int m = tid >> 2, q = tid & 3;                 // 512 uint4 (8KB) A tile
        async_b128(lds_off(&aQ[m][q * 4]), &ax[((size_t)slab * NROWS + row0 + m) * 4 + q]);
    }
    #pragma unroll
    for (int i = 0; i < 2; ++i) {                      // 1024 uint4 (16KB) B tile
        int p = tid + i * 512;
        int n = p >> 2, q = p & 3;
        async_b128(lds_off(&bQ[n][q * 4]), &bw[((size_t)slab * VPAD + colB + n) * 4 + q]);
    }
}

// Fallback path: convert fp32 -> bf16 on the fly (used when ws too small for the bf16 cache).
__device__ __forceinline__ void load_conv(unsigned (&aQ)[MT][LPAD],
                                          unsigned (&bQ)[256][LPAD],
                                          const float* __restrict__ x,
                                          const float* __restrict__ w,
                                          int row0, int colB, int k0, int tid) {
    #pragma unroll
    for (int i = 0; i < 4; ++i) {
        int p  = tid + i * 512;
        int m  = p >> 4;
        int kk = p & 15;
        const float2 v = *(const float2*)(x + (size_t)(row0 + m) * DDIM + k0 + 2 * kk);
        aQ[m][kk] = bf16pair(v.x, v.y);
    }
    #pragma unroll
    for (int i = 0; i < 8; ++i) {
        int p  = tid + i * 512;
        int kk = p >> 8;
        int n  = p & 255;
        int nc = colB + n;
        float w0 = 0.f, w1 = 0.f;
        if (nc < VDIM) {
            size_t b = (size_t)(k0 + 2 * kk) * VDIM + nc;
            w0 = w[b];
            w1 = w[b + VDIM];
        }
        bQ[n][kk] = bf16pair(w0, w1);
    }
}

// ---------------- phase 1: GEMM + per-col-block online stats ----------------
template <bool PACKED>
__global__ __launch_bounds__(512)
void ce_phase1(const float* __restrict__ x, const float* __restrict__ w,
               const uint4* __restrict__ ax, const uint4* __restrict__ bw,
               const int* __restrict__ tgt,
               float* __restrict__ ws_m, float* __restrict__ ws_s,
               float* __restrict__ tlogit) {
    __shared__ unsigned aQ[2][MT][LPAD];
    __shared__ unsigned bQ[2][256][LPAD];
    __shared__ float lmS[2][MT][4];
    __shared__ float lsS[2][MT][4];

    const int tid    = threadIdx.x;
    const int lane   = tid & 31;
    const int wv     = tid >> 5;
    const int rowgrp = wv & 1;          // 64-row group
    const int colgrp = wv >> 1;         // 0..7 -> 32-col group
    const int row0   = blockIdx.y * MT;
    const int colB   = blockIdx.x * 256;
    const int colWB  = colB + colgrp * 32;
    const int half   = lane >> 4;
    const int l15    = lane & 15;

    v8f acc[4][2];
    #pragma unroll
    for (int rs = 0; rs < 4; ++rs)
        #pragma unroll
        for (int cs = 0; cs < 2; ++cs) acc[rs][cs] = (v8f)0.0f;

    if constexpr (PACKED) {
        load_packed_async(aQ[0], bQ[0], ax, bw, row0, colB, 0, tid);
        wait_async0();
    } else {
        load_conv(aQ[0], bQ[0], x, w, row0, colB, 0, tid);
    }
    __syncthreads();

    for (int it = 0; it < KIT; ++it) {
        const int cur = it & 1;
        if (it + 1 < KIT) {
            if constexpr (PACKED) load_packed_async(aQ[cur ^ 1], bQ[cur ^ 1], ax, bw, row0, colB, it + 1, tid);
            else                  load_conv(aQ[cur ^ 1], bQ[cur ^ 1], x, w, row0, colB, (it + 1) * KS, tid);
        }

        Frag af[4];
        #pragma unroll
        for (int rs = 0; rs < 4; ++rs) {
            const unsigned* base = &aQ[cur][rowgrp * 64 + rs * 16 + l15][0];
            *(uint4*)&af[rs].u[0] = *(const uint4*)(base + half * 4);
            *(uint4*)&af[rs].u[4] = *(const uint4*)(base + 8 + half * 4);
        }
        Frag bfg[2];
        #pragma unroll
        for (int cs = 0; cs < 2; ++cs) {
            const unsigned* base = &bQ[cur][colgrp * 32 + cs * 16 + l15][0];
            *(uint4*)&bfg[cs].u[0] = *(const uint4*)(base + half * 4);
            *(uint4*)&bfg[cs].u[4] = *(const uint4*)(base + 8 + half * 4);
        }
        #pragma unroll
        for (int rs = 0; rs < 4; ++rs)
            #pragma unroll
            for (int cs = 0; cs < 2; ++cs)
                acc[rs][cs] = __builtin_amdgcn_wmma_f32_16x16x32_bf16(
                    false, af[rs].v, false, bfg[cs].v, (short)0, acc[rs][cs], false, false);

        if constexpr (PACKED) wait_async0();    // own prefetch copies into next buffer done
        __syncthreads();
    }

    // epilogue: per-row max/sumexp over this wave's 32 cols, combine 4 waves per 128-col block
    const int cbloc = colgrp >> 2;
    const int cg4   = colgrp & 3;

    #pragma unroll
    for (int rs = 0; rs < 4; ++rs) {
        #pragma unroll
        for (int r = 0; r < 8; ++r) {
            const int rloc = rowgrp * 64 + rs * 16 + r + half * 8;   // C layout: M = r + 8*half
            const int grow = row0 + rloc;

            float mx = -INFINITY;
            #pragma unroll
            for (int cs = 0; cs < 2; ++cs) {
                int n = colWB + cs * 16 + l15;
                float v = acc[rs][cs][r];
                if (n < VDIM && v > mx) mx = v;
            }
            #pragma unroll
            for (int off = 1; off <= 8; off <<= 1)
                mx = fmaxf(mx, __shfl_xor(mx, off, 32));

            float sm = 0.f;
            #pragma unroll
            for (int cs = 0; cs < 2; ++cs) {
                int n = colWB + cs * 16 + l15;
                if (n < VDIM) sm += expf(acc[rs][cs][r] - mx);
            }
            #pragma unroll
            for (int off = 1; off <= 8; off <<= 1)
                sm += __shfl_xor(sm, off, 32);

            if (l15 == 0) {
                lmS[cbloc][rloc][cg4] = mx;
                lsS[cbloc][rloc][cg4] = sm;
            }

            int tg = tgt[grow];
            int ct = tg < 0 ? 0 : (tg >= VDIM ? VDIM - 1 : tg);
            int tl = ct - colWB;
            if (tl >= 0 && tl < 32 && l15 == (tl & 15)) {
                int tc = tl >> 4;
                #pragma unroll
                for (int cs = 0; cs < 2; ++cs)
                    if (cs == tc) tlogit[grow] = acc[rs][cs][r];
            }
        }
    }
    __syncthreads();

    if (tid < 256) {
        const int rloc = tid & 127;
        const int cbl  = tid >> 7;
        const int cb   = blockIdx.x * 2 + cbl;
        if (cb < NUMCB) {
            float M = -INFINITY;
            #pragma unroll
            for (int j = 0; j < 4; ++j) M = fmaxf(M, lmS[cbl][rloc][j]);
            float S = 0.f;
            #pragma unroll
            for (int j = 0; j < 4; ++j) S += lsS[cbl][rloc][j] * expf(lmS[cbl][rloc][j] - M);
            ws_m[(size_t)cb * NROWS + row0 + rloc] = M;
            ws_s[(size_t)cb * NROWS + row0 + rloc] = S;
        }
    }
}

// ---------------- phases 2/3: deterministic logsumexp merge + reduce ----------------
__global__ __launch_bounds__(128)
void ce_phase2(const float* __restrict__ ws_m, const float* __restrict__ ws_s,
               const float* __restrict__ tlogit, const int* __restrict__ tgt,
               float* __restrict__ partials) {
    const int row = blockIdx.x * 128 + threadIdx.x;
    float M = -INFINITY, S = 0.f;
    for (int cb = 0; cb < NUMCB; ++cb) {
        float m = ws_m[(size_t)cb * NROWS + row];
        float s = ws_s[(size_t)cb * NROWS + row];
        float nM = fmaxf(M, m);
        S = S * expf(M - nM) + s * expf(m - nM);
        M = nM;
    }
    float lse  = M + logf(S);
    float loss = (tgt[row] != IGNORE_INDEX) ? (lse - tlogit[row]) : 0.f;

    __shared__ float red[128];
    red[threadIdx.x] = loss;
    __syncthreads();
    #pragma unroll
    for (int s = 64; s > 0; s >>= 1) {
        if (threadIdx.x < s) red[threadIdx.x] += red[threadIdx.x + s];
        __syncthreads();
    }
    if (threadIdx.x == 0) partials[blockIdx.x] = red[0];
}

__global__ __launch_bounds__(32)
void ce_phase3(const float* __restrict__ partials, float* __restrict__ out) {
    float v = partials[threadIdx.x];
    #pragma unroll
    for (int off = 16; off > 0; off >>= 1)
        v += __shfl_xor(v, off, 32);
    if (threadIdx.x == 0) out[0] = v;
}

extern "C" void kernel_launch(void* const* d_in, const int* in_sizes, int n_in,
                              void* d_out, int out_size, void* d_ws, size_t ws_size,
                              hipStream_t stream) {
    const float* x   = (const float*)d_in[0];   // [4096, 2048]
    const float* w   = (const float*)d_in[1];   // [2048, 50257]
    const int*   tgt = (const int*)d_in[2];     // [4096]

    float* ws   = (float*)d_ws;
    size_t cbN  = (size_t)NUMCB * NROWS;
    float* ws_m = ws;                       // NUMCB*N floats
    float* ws_s = ws + cbN;                 // NUMCB*N floats
    float* tlog = ws + 2 * cbN;             // N floats
    float* part = ws + 2 * cbN + NROWS;     // 32 floats
    size_t statf = 2 * cbN + NROWS + 32;    // floats used by stats (16B-aligned count)

    size_t bw_u4 = (size_t)KIT * VPAD * 4;  // packed W: 206.6 MB
    size_t ax_u4 = (size_t)KIT * NROWS * 4; // packed x: 16.8 MB
    size_t need  = statf * 4 + (bw_u4 + ax_u4) * 16;

    dim3 g1(GRIDX, NROWS / MT);             // (197, 32)
    if (ws_size >= need) {
        uint4* bw = (uint4*)(ws + statf);
        uint4* ax = bw + bw_u4;
        pack_w<<<dim3(VPAD / 256, KIT), 256, 0, stream>>>(w, bw);
        pack_x<<<(NROWS * KIT) / 256, 256, 0, stream>>>(x, ax);
        ce_phase1<true><<<g1, 512, 0, stream>>>(x, w, ax, bw, tgt, ws_m, ws_s, tlog);
    } else {
        ce_phase1<false><<<g1, 512, 0, stream>>>(x, w, nullptr, nullptr, tgt, ws_m, ws_s, tlog);
    }
    ce_phase2<<<NROWS / 128, 128, 0, stream>>>(ws_m, ws_s, tlog, tgt, part);
    ce_phase3<<<1, 32, 0, stream>>>(part, (float*)d_out);
}
